// Attention_71330816852838
// MI455X (gfx1250) — matrix operand order repaired and checked by reading, not compiled, not run on hardware
//
#include <hip/hip_runtime.h>
#include <hip/hip_fp16.h>

typedef _Float16 f16;
typedef __attribute__((ext_vector_type(16))) _Float16 v16h;
typedef __attribute__((ext_vector_type(8)))  float    v8f;

union U32H2 { unsigned int u; _Float16 h[2]; };

// ---------------------------------------------------------------------------
// WMMA helper: D = A(16x32 f16) * B(32x16 f16) + C(16x16 f32)
// ---------------------------------------------------------------------------
__device__ __forceinline__ v8f wmma_f16(v16h a, v16h b, v8f c) {
    return __builtin_amdgcn_wmma_f32_16x16x32_f16(
        /*neg_a=*/false, a, /*neg_b=*/false, b,
        /*c_mod=*/(short)0, c, /*reuse_a=*/false, /*reuse_b=*/false);
}

// ---------------------------------------------------------------------------
// A fragment (16x32, MxK), row-major source, lda elements/row.
// ISA layout: lane m=l&15, hi=l>>4; VGPR v<4: K=hi*8+2v; v>=4: K=16+hi*8+2(v-4)
// Per lane this is two contiguous 16-byte runs -> compiler emits b128 pairs.
// ---------------------------------------------------------------------------
__device__ __forceinline__ v16h load_a_frag(const f16* A, int lda) {
    int lane = threadIdx.x & 31;
    int m = lane & 15, hi = lane >> 4;
    const f16* p = A + (size_t)m * lda;
    v16h a;
#pragma unroll
    for (int v = 0; v < 4; ++v) {
        U32H2 t; t.u = *(const unsigned int*)(p + hi * 8 + 2 * v);
        a[2 * v] = t.h[0]; a[2 * v + 1] = t.h[1];
    }
#pragma unroll
    for (int v = 4; v < 8; ++v) {
        U32H2 t; t.u = *(const unsigned int*)(p + 16 + hi * 8 + 2 * (v - 4));
        a[2 * v] = t.h[0]; a[2 * v + 1] = t.h[1];
    }
    return a;
}

// ---------------------------------------------------------------------------
// B fragment (32x16, KxN) from row-major [K,N] storage (N contiguous), ldb.
// Lane: col n=l&15, hi=l>>4; element s<8: K=hi*8+s; s>=8: K=16+hi*8+(s-8).
// ---------------------------------------------------------------------------
__device__ __forceinline__ v16h load_b_frag(const f16* B, int ldb) {
    int lane = threadIdx.x & 31;
    int n = lane & 15, hi = lane >> 4;
    const f16* p = B + n;
    v16h b;
#pragma unroll
    for (int s = 0; s < 8; ++s) {
        b[s]     = p[(size_t)(hi * 8 + s) * ldb];
        b[s + 8] = p[(size_t)(16 + hi * 8 + s) * ldb];
    }
    return b;
}

// ---------------------------------------------------------------------------
// B fragment for Q*K^T: logical B[kd, n] = Krows[n*ldm + kd] (transposed view).
// Lane: col n=l&15, hi=l>>4; element s<8: kd=hi*8+s; s>=8: kd=16+hi*8+(s-8).
// ---------------------------------------------------------------------------
__device__ __forceinline__ v16h load_bt_frag(const f16* M, int ldm) {
    int lane = threadIdx.x & 31;
    int n = lane & 15, hi = lane >> 4;
    const f16* p = M + (size_t)n * ldm;
    v16h b;
#pragma unroll
    for (int v = 0; v < 4; ++v) {
        U32H2 t; t.u = *(const unsigned int*)(p + hi * 8 + 2 * v);
        b[2 * v] = t.h[0]; b[2 * v + 1] = t.h[1];
        U32H2 u; u.u = *(const unsigned int*)(p + 16 + hi * 8 + 2 * v);
        b[2 * v + 8] = u.h[0]; b[2 * v + 9] = u.h[1];
    }
    return b;
}

// ---------------------------------------------------------------------------
// Elementwise f32 -> f16 cast
// ---------------------------------------------------------------------------
__global__ void k_cast_f16(const float* __restrict__ in, f16* __restrict__ out, int n) {
    int i = blockIdx.x * blockDim.x + threadIdx.x;
    if (i < n) out[i] = (f16)in[i];
}

// in: [rows, cols] f32 row-major  ->  out: [cols, rows] f16 (transposed)
__global__ void k_transpose_cast(const float* __restrict__ in, f16* __restrict__ out,
                                 int rows, int cols) {
    int i = blockIdx.x * blockDim.x + threadIdx.x;
    if (i < rows * cols) {
        int r = i / cols, c = i % cols;
        out[(size_t)c * rows + r] = (f16)in[i];
    }
}

// ---------------------------------------------------------------------------
// Wave-tile GEMM: each wave computes a 32x64 tile of C = A[MxK] * B[KxN].
// Two A fragments share four B fragments -> 8 WMMAs per 32-deep K step.
// Software-pipelined: next iteration's fragments are fetched before issuing
// the current WMMA burst, so s_wait_loadcnt overlaps with matrix math.
// A row-major (lda), B stored [K,N] row-major (ldb). Output f16 or f32+bias.
// All branches wave-uniform -> EXEC all-ones at every WMMA.
// ---------------------------------------------------------------------------
__global__ void k_gemm32x64(const f16* __restrict__ A, const f16* __restrict__ B,
                            f16* __restrict__ Ch, float* __restrict__ Cf,
                            const float* __restrict__ bias,
                            int M, int N, int K, int lda, int ldb, int ldc) {
    int wave = (blockIdx.x * blockDim.x + threadIdx.x) >> 5;
    int lane = threadIdx.x & 31;
    int nblk = N >> 6;
    int mt = wave / nblk;
    int nt = wave % nblk;
    if (mt * 32 >= M) return;

    const f16* Ap = A + (size_t)(mt * 32) * lda;
    const f16* Bp = B + (size_t)(nt * 64);

    v8f acc[8] = {};

    // Prologue: fragments for k=0.
    v16h a0 = load_a_frag(Ap, lda);
    v16h a1 = load_a_frag(Ap + (size_t)16 * lda, lda);
    v16h b0 = load_b_frag(Bp,      ldb);
    v16h b1 = load_b_frag(Bp + 16, ldb);
    v16h b2 = load_b_frag(Bp + 32, ldb);
    v16h b3 = load_b_frag(Bp + 48, ldb);

    int k0 = 0;
    for (; k0 + 32 < K; k0 += 32) {
        // Speculative prefetch two blocks ahead (safe: dropped on xlat fault).
        __builtin_prefetch(Ap + k0 + 64, 0, 1);
        __builtin_prefetch(Bp + (size_t)(k0 + 64) * ldb, 0, 1);

        // Fetch next K block before consuming the current one.
        const f16* An = Ap + (k0 + 32);
        const f16* Bn = Bp + (size_t)(k0 + 32) * ldb;
        v16h na0 = load_a_frag(An, lda);
        v16h na1 = load_a_frag(An + (size_t)16 * lda, lda);
        v16h nb0 = load_b_frag(Bn,      ldb);
        v16h nb1 = load_b_frag(Bn + 16, ldb);
        v16h nb2 = load_b_frag(Bn + 32, ldb);
        v16h nb3 = load_b_frag(Bn + 48, ldb);

        acc[0] = wmma_f16(a0, b0, acc[0]);
        acc[1] = wmma_f16(a0, b1, acc[1]);
        acc[2] = wmma_f16(a0, b2, acc[2]);
        acc[3] = wmma_f16(a0, b3, acc[3]);
        acc[4] = wmma_f16(a1, b0, acc[4]);
        acc[5] = wmma_f16(a1, b1, acc[5]);
        acc[6] = wmma_f16(a1, b2, acc[6]);
        acc[7] = wmma_f16(a1, b3, acc[7]);

        a0 = na0; a1 = na1;
        b0 = nb0; b1 = nb1; b2 = nb2; b3 = nb3;
    }
    // Epilogue: last K block.
    acc[0] = wmma_f16(a0, b0, acc[0]);
    acc[1] = wmma_f16(a0, b1, acc[1]);
    acc[2] = wmma_f16(a0, b2, acc[2]);
    acc[3] = wmma_f16(a0, b3, acc[3]);
    acc[4] = wmma_f16(a1, b0, acc[4]);
    acc[5] = wmma_f16(a1, b1, acc[5]);
    acc[6] = wmma_f16(a1, b2, acc[6]);
    acc[7] = wmma_f16(a1, b3, acc[7]);

    int hi = lane >> 4, lm = lane & 15;
#pragma unroll
    for (int mi = 0; mi < 2; ++mi) {
        int rowb = mt * 32 + mi * 16 + 8 * hi;
#pragma unroll
        for (int j = 0; j < 4; ++j) {
            int col = nt * 64 + 16 * j + lm;
#pragma unroll
            for (int v = 0; v < 8; ++v) {
                float val = acc[mi * 4 + j][v];
                size_t idx = (size_t)(rowb + v) * ldc + col;
                if (Cf) Cf[idx] = val + (bias ? bias[col] : 0.0f);
                else    Ch[idx] = (f16)val;
            }
        }
    }
}

// ---------------------------------------------------------------------------
// Flash attention: one wave per (b, h, 32-query tile). Online softmax over
// 32-key blocks; QK^T and PV via WMMA. Every K^T / V fragment is reused by
// both query sub-tiles (16 WMMAs per 32-key block).
// qkv: [4096, 3072] f16 (q|k|v concat).  O: [4096, 1024] f16, heads merged.
// ---------------------------------------------------------------------------
#define SEQ 2048
__global__ void k_flash_attn(const f16* __restrict__ qkv, f16* __restrict__ O) {
    __shared__ f16 pstage[8][32 * 32];   // per-wave P staging (2 KB each)

    int wave = (blockIdx.x * blockDim.x + threadIdx.x) >> 5;
    int lane = threadIdx.x & 31;
    int wib  = (threadIdx.x >> 5) & 7;
    int hi   = lane >> 4;
    int lm   = lane & 15;

    const int qtiles = SEQ / 32;           // 64
    int bh = wave / qtiles;
    int qt = wave % qtiles;
    if (bh >= 32) return;                  // wave-uniform
    int b = bh >> 4, h = bh & 15;

    size_t rowbase = (size_t)b * SEQ;
    const f16* Q  = qkv + (rowbase + (size_t)qt * 32) * 3072 + h * 64;
    const f16* Kp = qkv + rowbase * 3072 + 1024 + h * 64;
    const f16* Vp = qkv + rowbase * 3072 + 2048 + h * 64;

    // Q fragments: [mi][kh] = rows 16*mi..16*mi+15, head-dim half kh.
    v16h qa[2][2];
#pragma unroll
    for (int mi = 0; mi < 2; ++mi) {
        qa[mi][0] = load_a_frag(Q + (size_t)(16 * mi) * 3072,      3072);
        qa[mi][1] = load_a_frag(Q + (size_t)(16 * mi) * 3072 + 32, 3072);
    }

    float mrow[2][8], lrow[2][8];
#pragma unroll
    for (int mi = 0; mi < 2; ++mi)
#pragma unroll
        for (int v = 0; v < 8; ++v) { mrow[mi][v] = -3.0e38f; lrow[mi][v] = 0.0f; }
    v8f oacc[2][4] = {};

    f16* pb = &pstage[wib][0];
    const float scale = 0.125f;            // 1/sqrt(64)

    for (int kb = 0; kb < SEQ; kb += 32) {
        const f16* Kt = Kp + (size_t)kb * 3072;

        // K^T fragments: [ni][kh] = keys 16*ni.., head-dim half kh.
        v16h bt[2][2];
        bt[0][0] = load_bt_frag(Kt,                  3072);
        bt[0][1] = load_bt_frag(Kt + 32,             3072);
        bt[1][0] = load_bt_frag(Kt + 16 * 3072,      3072);
        bt[1][1] = load_bt_frag(Kt + 16 * 3072 + 32, 3072);

        // S = Q K^T : 8 WMMAs, full A/B fragment reuse.
        v8f s[2][2] = {};
#pragma unroll
        for (int mi = 0; mi < 2; ++mi)
#pragma unroll
            for (int ni = 0; ni < 2; ++ni) {
                s[mi][ni] = wmma_f16(qa[mi][0], bt[ni][0], s[mi][ni]);
                s[mi][ni] = wmma_f16(qa[mi][1], bt[ni][1], s[mi][ni]);
            }

        // Online softmax per query sub-tile.
#pragma unroll
        for (int mi = 0; mi < 2; ++mi) {
#pragma unroll
            for (int v = 0; v < 8; ++v) {
                float a0 = s[mi][0][v] * scale;
                float a1 = s[mi][1][v] * scale;
                float t = fmaxf(a0, a1);
                t = fmaxf(t, __shfl_xor(t, 1));
                t = fmaxf(t, __shfl_xor(t, 2));
                t = fmaxf(t, __shfl_xor(t, 4));
                t = fmaxf(t, __shfl_xor(t, 8));
                float mn = fmaxf(mrow[mi][v], t);
                float al = __expf(mrow[mi][v] - mn);
                float p0 = __expf(a0 - mn);
                float p1 = __expf(a1 - mn);
                float rs = p0 + p1;
                rs += __shfl_xor(rs, 1);
                rs += __shfl_xor(rs, 2);
                rs += __shfl_xor(rs, 4);
                rs += __shfl_xor(rs, 8);
                lrow[mi][v] = lrow[mi][v] * al + rs;
                mrow[mi][v] = mn;
                oacc[mi][0][v] *= al; oacc[mi][1][v] *= al;
                oacc[mi][2][v] *= al; oacc[mi][3][v] *= al;
                int r = v + 8 * hi;
                pb[mi * 512 + r * 32 + lm]      = (f16)p0;
                pb[mi * 512 + r * 32 + 16 + lm] = (f16)p1;
            }
        }
        asm volatile("s_wait_dscnt 0" ::: "memory");

        // PV: V fragments loaded once, reused by both query sub-tiles.
        v16h pa0 = load_a_frag(pb,       32);
        v16h pa1 = load_a_frag(pb + 512, 32);
        const f16* Vt = Vp + (size_t)kb * 3072;
#pragma unroll
        for (int j = 0; j < 4; ++j) {
            v16h bv = load_b_frag(Vt + 16 * j, 3072);
            oacc[0][j] = wmma_f16(pa0, bv, oacc[0][j]);
            oacc[1][j] = wmma_f16(pa1, bv, oacc[1][j]);
        }
    }

    // Normalize and write O tile into merged [B*S, D] f16 layout.
    int col0 = h * 64 + lm;
#pragma unroll
    for (int mi = 0; mi < 2; ++mi) {
        size_t row0 = rowbase + (size_t)qt * 32 + 16 * mi + 8 * hi;
#pragma unroll
        for (int v = 0; v < 8; ++v) {
            float inv = 1.0f / lrow[mi][v];
#pragma unroll
            for (int j = 0; j < 4; ++j) {
                O[(row0 + v) * 1024 + col0 + 16 * j] = (f16)(oacc[mi][j][v] * inv);
            }
        }
    }
}

// ---------------------------------------------------------------------------
// Host-side launcher
// ---------------------------------------------------------------------------
extern "C" void kernel_launch(void* const* d_in, const int* in_sizes, int n_in,
                              void* d_out, int out_size, void* d_ws, size_t ws_size,
                              hipStream_t stream) {
    const float* x      = (const float*)d_in[0];   // [2,2048,1024]
    const float* w_qkv  = (const float*)d_in[1];   // [3072,1024]
    const float* w_proj = (const float*)d_in[2];   // [1024,1024]
    const float* b_proj = (const float*)d_in[3];   // [1024]
    float* out = (float*)d_out;                    // [2,2048,1024] f32

    char* ws = (char*)d_ws;
    // Workspace layout (bytes):
    f16* x16    = (f16*)(ws);                              //  8 MB: [4096,1024]
    f16* wqkvT  = (f16*)(ws + 8388608);                    //  6 MB: [1024,3072]
    f16* wprojT = (f16*)(ws + 8388608 + 6291456);          //  2 MB: [1024,1024]
    f16* qkv16  = (f16*)(ws + 16777216);                   // 24 MB: [4096,3072]
    f16* attn16 = (f16*)(ws + 16777216 + 25165824);        //  8 MB: [4096,1024]

    // 1) precision conversion (+ weight transpose so GEMM B is [K,N])
    k_cast_f16      <<<16384, 256, 0, stream>>>(x, x16, 4096 * 1024);
    k_transpose_cast<<<12288, 256, 0, stream>>>(w_qkv, wqkvT, 3072, 1024);
    k_transpose_cast<<< 4096, 256, 0, stream>>>(w_proj, wprojT, 1024, 1024);

    // 2) QKV projection: [4096,1024] x [1024,3072] -> [4096,3072] f16
    //    waves = 128 Mtiles(32) * 48 Nblocks(64) = 6144 -> 768 blocks
    k_gemm32x64<<<768, 256, 0, stream>>>(x16, wqkvT, qkv16, nullptr, nullptr,
                                         4096, 3072, 1024, 1024, 3072, 3072);

    // 3) flash attention: 2*16*64 = 2048 waves -> 256 blocks of 256 thr
    k_flash_attn<<<256, 256, 0, stream>>>(qkv16, attn16);

    // 4) output projection + bias: [4096,1024] x [1024,1024] -> f32 out
    //    waves = 128 * 16 = 2048 -> 256 blocks
    k_gemm32x64<<<256, 256, 0, stream>>>(attn16, wprojT, nullptr, out, b_proj,
                                         4096, 1024, 1024, 1024, 1024, 1024);
}
